// AttnBlock_70686571758041
// MI455X (gfx1250) — matrix operand order repaired and checked by reading, not compiled, not run on hardware
//
#include <hip/hip_runtime.h>

// ---------------------------------------------------------------------------
// AttnBlock for MI455X (gfx1250, wave32, WMMA + Tensor Data Mover)
// B=32, C=256, H=W=32 (N=1024)
// Pipeline: cvt weights -> GroupNorm(h^T bf16) -> q/k/v proj (WMMA bf16, TDM
//           double-buffered LDS staging) -> flash attention (WMMA bf16, TDM)
//           -> out proj + residual (WMMA bf16, f32 out)
// ---------------------------------------------------------------------------

typedef __bf16 bf16;
typedef __attribute__((ext_vector_type(16))) __bf16 v16bf;
typedef __attribute__((ext_vector_type(8)))  __bf16 v8bf;
typedef __attribute__((ext_vector_type(8)))  float  v8f;

#define BATCH 32
#define CCH   256
#define NSP   1024            // H*W
#define NC    (NSP * CCH)     // 262144 elements per batch image
#define LDA   40              // padded K-stride (bf16): 64B data + 16B pad
#define LDSS  72              // padded stride for 64-wide S / P tiles

// ---------------------------------------------------------------------------
// Tensor Data Mover: hardware 2D tile DMA global->LDS with row padding that
// reproduces the LDA=40 layout (pad_interval=3 -> every 16 DWORDs, pad_amount
// =3 -> 4 DWORDs pad). Issued by one wave, tracked with TENSORcnt.
// ---------------------------------------------------------------------------
#if defined(__has_builtin)
#  if __has_builtin(__builtin_amdgcn_tensor_load_to_lds) && \
      __has_builtin(__builtin_amdgcn_s_wait_tensorcnt)
#    define HAVE_TDM 1
#  endif
#endif
#ifndef HAVE_TDM
#  define HAVE_TDM 0
#endif

#if HAVE_TDM
typedef __attribute__((ext_vector_type(4))) unsigned int u32x4;
typedef __attribute__((ext_vector_type(4))) int i32x4;
typedef __attribute__((ext_vector_type(8))) int i32x8;

// Load tile_d1 rows x tile_d0 bf16 elements (row pitch stride0 elements in
// global memory) into LDS at lds_dst, 80B-padded rows when tile_d0 == 32.
static __device__ inline void tdm_load_tile_2d(void* lds_dst, const bf16* gsrc,
                                               unsigned tile_d0,
                                               unsigned tile_d1,
                                               unsigned long long stride0) {
  unsigned long long ga = (unsigned long long)(uintptr_t)gsrc;
  u32x4 g0;
  g0[0] = 1u;                                   // count=1 (valid user D#)
  g0[1] = (unsigned)(uintptr_t)lds_dst;         // lds_addr (byte offset)
  g0[2] = (unsigned)ga;                         // global_addr[31:0]
  g0[3] = (unsigned)(ga >> 32) | (2u << 30);    // global_addr[56:32] | type=2
  i32x8 g1;
  // wg_mask=0 | data_size=1(2B)<<16 | pad_enable<<20 | pad_interval=3<<22
  // (pad after 16 DWORDs = 64B) | pad_amount=3<<25 (4 DWORDs = 16B)
  g1[0] = (int)((1u << 16) | (1u << 20) | (3u << 22) | (3u << 25));
  g1[1] = (int)((tile_d0 & 0xFFFFu) << 16);     // atomic_addr=0 | td0[15:0]
  g1[2] = (int)((tile_d0 >> 16) | ((tile_d1 & 0xFFFFu) << 16));
  g1[3] = (int)((tile_d1 >> 16) | (tile_d0 << 16));   // td1[31:16]|tile_dim0
  g1[4] = (int)(tile_d1 & 0xFFFFu);             // tile_dim1 | tile_dim2=0
  g1[5] = (int)(unsigned)stride0;               // tensor_dim0_stride[31:0]
  g1[6] = (int)((unsigned)(stride0 >> 32) & 0xFFFFu);
  g1[7] = 0;
  const i32x4 z4 = {0, 0, 0, 0};
#if __clang_major__ >= 23
  const i32x8 z8 = {0, 0, 0, 0, 0, 0, 0, 0};
  __builtin_amdgcn_tensor_load_to_lds(g0, g1, z4, z4, z8, 0);
#else
  __builtin_amdgcn_tensor_load_to_lds(g0, g1, z4, z4, 0);
#endif
}
#endif  // HAVE_TDM

union FragU { v16bf v; v8bf h[2]; };

// A fragment (16 rows x 32 K), source row-major [row][ld] bf16, K-contiguous.
static __device__ inline v16bf load_frag_a(const bf16* src, int ld, int lane) {
  int r = lane & 15, hi = lane >> 4;
  FragU f;
  f.h[0] = *(const v8bf*)(src + r * ld + hi * 8);
  f.h[1] = *(const v8bf*)(src + r * ld + 16 + hi * 8);
  return f.v;
}

// B fragment (32 K x 16 cols), source stored as B^T row-major [col][ld].
static __device__ inline v16bf load_frag_b(const bf16* srcT, int ld, int lane) {
  int c = lane & 15, hi = lane >> 4;
  FragU f;
  f.h[0] = *(const v8bf*)(srcT + c * ld + hi * 8);
  f.h[1] = *(const v8bf*)(srcT + c * ld + 16 + hi * 8);
  return f.v;
}

static __device__ inline v8f wmma_bf16(v16bf a, v16bf b, v8f c) {
  return __builtin_amdgcn_wmma_f32_16x16x32_bf16(false, a, false, b,
                                                 (short)0, c, false, false);
}

// ---------------------------------------------------------------------------
// Kernel 0: convert wq,wk,wv,wo (f32 [256][256]) to bf16
// ---------------------------------------------------------------------------
__global__ __launch_bounds__(256) void cvt_w_kernel(
    const float* __restrict__ wq, const float* __restrict__ wk,
    const float* __restrict__ wv, const float* __restrict__ wo,
    bf16* __restrict__ out) {
  int i = blockIdx.x * 256 + threadIdx.x;   // 4*65536 total
  int m = i >> 16, r = i & 65535;
  float v = (m == 0) ? wq[r] : (m == 1) ? wk[r] : (m == 2) ? wv[r] : wo[r];
  out[i] = (bf16)v;
}

// ---------------------------------------------------------------------------
// Kernel 1: GroupNorm (32 groups of 8 channels) -> h^T [B][N][C] bf16
// ---------------------------------------------------------------------------
__global__ __launch_bounds__(256) void gn_kernel(
    const float* __restrict__ x, const float* __restrict__ gs,
    const float* __restrict__ gb, bf16* __restrict__ hT) {
  __shared__ float sx[8 * 1024];
  __shared__ float rA[256], rB[256];
  const int tid = threadIdx.x;
  const int b = blockIdx.x >> 5, g = blockIdx.x & 31;
  const float* xp = x + ((size_t)b * CCH + g * 8) * NSP;

  float s = 0.f, ss = 0.f;
  for (int i = tid; i < 8192; i += 256) {
    float v = xp[i];
    sx[i] = v; s += v; ss += v * v;
  }
  rA[tid] = s; rB[tid] = ss;
  __syncthreads();
  for (int off = 128; off > 0; off >>= 1) {
    if (tid < off) { rA[tid] += rA[tid + off]; rB[tid] += rB[tid + off]; }
    __syncthreads();
  }
  const float mean = rA[0] * (1.f / 8192.f);
  const float var  = rB[0] * (1.f / 8192.f) - mean * mean;
  const float rstd = rsqrtf(var + 1e-5f);

  for (int n = tid; n < NSP; n += 256) {
    v8bf tmp;
    #pragma unroll
    for (int cc = 0; cc < 8; ++cc) {
      int c = g * 8 + cc;
      float v = (sx[cc * 1024 + n] - mean) * rstd;
      tmp[cc] = (bf16)(v * gs[c] + gb[c]);
    }
    *(v8bf*)(hT + ((size_t)b * NSP + n) * CCH + g * 8) = tmp;
  }
}

// ---------------------------------------------------------------------------
// Shared GEMM core: 128(M) x 64(N) block tile, K=256 in 32-chunks, 8 waves of
// 32x32, TDM double-buffered LDS staging. A [M][K] K-major, B^T [N][K].
// ---------------------------------------------------------------------------
#define GEMM_CORE(A_PTR, B_PTR)                                               \
  __shared__ __attribute__((aligned(16))) bf16 sA[2][128 * LDA];              \
  __shared__ __attribute__((aligned(16))) bf16 sB[2][64 * LDA];               \
  const int tid = threadIdx.x;                                                \
  const int lane = tid & 31, wave = tid >> 5;                                 \
  const int col = lane & 15, rhi = lane >> 4;                                 \
  const v8f vzero = {0.f, 0.f, 0.f, 0.f, 0.f, 0.f, 0.f, 0.f};                 \
  v8f acc[4] = {vzero, vzero, vzero, vzero};                                  \
  TDM_PROLOGUE(A_PTR, B_PTR)                                                  \
  for (int kc = 0; kc < 8; ++kc) {                                            \
    const int k0 = kc * 32;                                                   \
    const int cur = kc & 1;                                                   \
    STAGE_AB(A_PTR, B_PTR)                                                    \
    __syncthreads();                                                          \
    v16bf a0 = load_frag_a(sA[cur] + ((wave & 3) * 32 + 0)  * LDA, LDA, lane);\
    v16bf a1 = load_frag_a(sA[cur] + ((wave & 3) * 32 + 16) * LDA, LDA, lane);\
    v16bf b0 = load_frag_b(sB[cur] + ((wave >> 2) * 32 + 0)  * LDA, LDA, lane);\
    v16bf b1 = load_frag_b(sB[cur] + ((wave >> 2) * 32 + 16) * LDA, LDA, lane);\
    acc[0] = wmma_bf16(a0, b0, acc[0]);                                       \
    acc[1] = wmma_bf16(a0, b1, acc[1]);                                       \
    acc[2] = wmma_bf16(a1, b0, acc[2]);                                       \
    acc[3] = wmma_bf16(a1, b1, acc[3]);                                       \
    __syncthreads();                                                          \
  }

#if HAVE_TDM
#define TDM_PROLOGUE(A_PTR, B_PTR)                                            \
  if (wave == 0) {                                                            \
    tdm_load_tile_2d(sA[0], (A_PTR) + m0 * CCH, 32, 128, CCH);                \
    tdm_load_tile_2d(sB[0], (B_PTR) + (size_t)n0 * CCH, 32, 64, CCH);         \
  }
#define STAGE_AB(A_PTR, B_PTR)                                                \
  if (wave == 0) {                                                            \
    if (kc < 7) { /* prefetch tile k+1 into alternate buffer via TDM */       \
      tdm_load_tile_2d(sA[cur ^ 1], (A_PTR) + m0 * CCH + k0 + 32,             \
                       32, 128, CCH);                                         \
      tdm_load_tile_2d(sB[cur ^ 1], (B_PTR) + (size_t)n0 * CCH + k0 + 32,     \
                       32, 64, CCH);                                          \
      __builtin_amdgcn_s_wait_tensorcnt(2); /* oldest 2 (tile k) landed */    \
    } else {                                                                  \
      __builtin_amdgcn_s_wait_tensorcnt(0);                                   \
    }                                                                         \
  }
#else
#define TDM_PROLOGUE(A_PTR, B_PTR)
#define STAGE_AB(A_PTR, B_PTR)                                                \
  {                                                                           \
    _Pragma("unroll")                                                         \
    for (int i = 0; i < 2; ++i) {                                             \
      int idx = tid + i * 256;                                                \
      int r = idx >> 2, o = (idx & 3) * 8;                                    \
      *(v8bf*)(sA[cur] + r * LDA + o) =                                       \
          *(const v8bf*)((A_PTR) + (m0 + r) * CCH + k0 + o);                  \
    }                                                                         \
    int r = tid >> 2, o = (tid & 3) * 8;                                      \
    *(v8bf*)(sB[cur] + r * LDA + o) =                                         \
        *(const v8bf*)((B_PTR) + (size_t)(n0 + r) * CCH + k0 + o);            \
  }
#endif

// ---------------------------------------------------------------------------
// Kernel 2: q/k/v projections. mode (blockIdx.z): 0=q -> q^T[N][C] * 1/16,
// 1=k -> k^T[N][C], 2=v -> v[C][N].
// ---------------------------------------------------------------------------
__global__ __launch_bounds__(256) void proj_gemm_kernel(
    const bf16* __restrict__ wbf, const float* __restrict__ bq,
    const float* __restrict__ bk, const float* __restrict__ bv,
    const bf16* __restrict__ hT, bf16* __restrict__ qT,
    bf16* __restrict__ kT, bf16* __restrict__ vv) {
  const int mode = blockIdx.z;
  const bf16* A = wbf + (size_t)mode * 65536;
  const float* bias = (mode == 0) ? bq : (mode == 1) ? bk : bv;
  const int b  = blockIdx.y >> 1;
  const int m0 = (blockIdx.y & 1) * 128;
  const int n0 = blockIdx.x * 64;
  const bf16* Bm = hT + (size_t)b * NC;

  GEMM_CORE(A, Bm)

  #pragma unroll
  for (int t = 0; t < 4; ++t) {
    int rb = t >> 1, cb = t & 1;
    int mBase = m0 + (wave & 3) * 32 + rb * 16 + rhi * 8;
    int nn = n0 + (wave >> 2) * 32 + cb * 16 + col;
    #pragma unroll
    for (int i = 0; i < 8; ++i) {
      int mm = mBase + i;                        // output channel
      float val = acc[t][i] + bias[mm];
      if (mode == 0)
        qT[(size_t)b * NC + (size_t)nn * CCH + mm] = (bf16)(val * 0.0625f);
      else if (mode == 1)
        kT[(size_t)b * NC + (size_t)nn * CCH + mm] = (bf16)val;
      else
        vv[(size_t)b * NC + (size_t)mm * NSP + nn] = (bf16)val;
    }
  }
}

// ---------------------------------------------------------------------------
// Kernel 4: out projection + residual: out[b][o][n] = x + Wo . o_attn + bo.
// ---------------------------------------------------------------------------
__global__ __launch_bounds__(256) void outproj_gemm_kernel(
    const bf16* __restrict__ woBf, const float* __restrict__ bo,
    const bf16* __restrict__ oT, const float* __restrict__ x,
    float* __restrict__ out) {
  const int b  = blockIdx.y >> 1;
  const int m0 = (blockIdx.y & 1) * 128;
  const int n0 = blockIdx.x * 64;
  const bf16* Bm = oT + (size_t)b * NC;

  GEMM_CORE(woBf, Bm)

  #pragma unroll
  for (int t = 0; t < 4; ++t) {
    int rb = t >> 1, cb = t & 1;
    int mBase = m0 + (wave & 3) * 32 + rb * 16 + rhi * 8;
    int nn = n0 + (wave >> 2) * 32 + cb * 16 + col;
    #pragma unroll
    for (int i = 0; i < 8; ++i) {
      int mm = mBase + i;
      size_t idx = (size_t)b * NC + (size_t)mm * NSP + nn;
      out[idx] = x[idx] + acc[t][i] + bo[mm];
    }
  }
}

// ---------------------------------------------------------------------------
// Kernel 3: flash attention. Block = 64 query rows of one batch; 16 m-tiles
// of 64 keys: S = q^T k (WMMA), online softmax in LDS, O += P * v^T (WMMA).
// TDM stages Q/K/V tiles.
// ---------------------------------------------------------------------------
__global__ __launch_bounds__(256) void attn_kernel(
    const bf16* __restrict__ qT, const bf16* __restrict__ kT,
    const bf16* __restrict__ vv, bf16* __restrict__ oT) {
  __shared__ __attribute__((aligned(16))) bf16  sQ[64 * LDA];
  __shared__ __attribute__((aligned(16))) bf16  sK[64 * LDA];
  __shared__ __attribute__((aligned(16))) bf16  sV[256 * LDA];
  __shared__ __attribute__((aligned(16))) float sS[64 * LDSS];
  __shared__ __attribute__((aligned(16))) bf16  sP[64 * LDSS];
  __shared__ float sM[64], sL[64], sAlpha[64];
  __shared__ float sRed[64 * 4], sRed2[64 * 4];

  const int tid = threadIdx.x;
  const int lane = tid & 31, wave = tid >> 5;
  const int col = lane & 15, rhi = lane >> 4;
  const int b = blockIdx.y;
  const int n0 = blockIdx.x * 64;
  const size_t bO = (size_t)b * NC;
  const int sr = tid >> 2, so = (tid & 3) * 8;   // fallback staging map

  if (tid < 64) { sM[tid] = -1e30f; sL[tid] = 0.f; }
  __syncthreads();

  const v8f vzero = {0.f, 0.f, 0.f, 0.f, 0.f, 0.f, 0.f, 0.f};
  v8f accO[8];
  #pragma unroll
  for (int j = 0; j < 8; ++j) accO[j] = vzero;

  for (int mt = 0; mt < 16; ++mt) {
    const int m0 = mt * 64;
    v8f accS[2] = {vzero, vzero};

    // ---- S = q[n0:n0+64,:] . k[m0:m0+64,:]^T, K = 256 in 32-chunks
    for (int kc = 0; kc < 8; ++kc) {
      const int k0 = kc * 32;
#if HAVE_TDM
      if (wave == 0) {
        tdm_load_tile_2d(sQ, qT + bO + (size_t)n0 * CCH + k0, 32, 64, CCH);
        tdm_load_tile_2d(sK, kT + bO + (size_t)m0 * CCH + k0, 32, 64, CCH);
        __builtin_amdgcn_s_wait_tensorcnt(0);
      }
#else
      *(v8bf*)(sQ + sr * LDA + so) =
          *(const v8bf*)(qT + bO + (size_t)(n0 + sr) * CCH + k0 + so);
      *(v8bf*)(sK + sr * LDA + so) =
          *(const v8bf*)(kT + bO + (size_t)(m0 + sr) * CCH + k0 + so);
#endif
      __syncthreads();
      v16bf aQ  = load_frag_a(sQ + (wave & 3) * 16 * LDA, LDA, lane);
      v16bf bK0 = load_frag_b(sK + ((wave >> 2) * 32 + 0)  * LDA, LDA, lane);
      v16bf bK1 = load_frag_b(sK + ((wave >> 2) * 32 + 16) * LDA, LDA, lane);
      accS[0] = wmma_bf16(aQ, bK0, accS[0]);
      accS[1] = wmma_bf16(aQ, bK1, accS[1]);
      __syncthreads();
    }
    // ---- spill S tiles to LDS
    #pragma unroll
    for (int t = 0; t < 2; ++t)
      #pragma unroll
      for (int i = 0; i < 8; ++i)
        sS[((wave & 3) * 16 + rhi * 8 + i) * LDSS +
           (wave >> 2) * 32 + t * 16 + col] = accS[t][i];
    __syncthreads();

    // ---- online softmax: 4 threads per row, 16 cols each
    {
      const int r = tid >> 2, seg = tid & 3;
      float lmax = -1e30f;
      #pragma unroll
      for (int j = 0; j < 16; ++j)
        lmax = fmaxf(lmax, sS[r * LDSS + seg * 16 + j]);
      sRed[r * 4 + seg] = lmax;
      __syncthreads();
      float tmax = fmaxf(fmaxf(sRed[r * 4 + 0], sRed[r * 4 + 1]),
                         fmaxf(sRed[r * 4 + 2], sRed[r * 4 + 3]));
      float mold = sM[r];
      float mnew = fmaxf(mold, tmax);
      float alpha = __expf(mold - mnew);
      float lsum = 0.f;
      #pragma unroll
      for (int j = 0; j < 16; ++j) {
        int c = seg * 16 + j;
        float p = __expf(sS[r * LDSS + c] - mnew);
        sP[r * LDSS + c] = (bf16)p;
        lsum += p;
      }
      sRed2[r * 4 + seg] = lsum;
      __syncthreads();
      if (seg == 0) {
        float rs = sRed2[r * 4 + 0] + sRed2[r * 4 + 1] +
                   sRed2[r * 4 + 2] + sRed2[r * 4 + 3];
        sL[r] = sL[r] * alpha + rs;
        sM[r] = mnew;
        sAlpha[r] = alpha;
      }
    }
    __syncthreads();

    // ---- rescale O accumulators, then O += P(64x64) * v^T(64x256)
    float alphas[8];
    #pragma unroll
    for (int i = 0; i < 8; ++i)
      alphas[i] = sAlpha[(wave & 3) * 16 + rhi * 8 + i];
    #pragma unroll
    for (int j = 0; j < 8; ++j)
      #pragma unroll
      for (int i = 0; i < 8; ++i)
        accO[j][i] *= alphas[i];

    for (int kc2 = 0; kc2 < 2; ++kc2) {
#if HAVE_TDM
      if (wave == 0) {   // v chunk: 256 c-rows x 32 m
        tdm_load_tile_2d(sV, vv + bO + m0 + kc2 * 32, 32, 256, NSP);
        __builtin_amdgcn_s_wait_tensorcnt(0);
      }
#else
      {
        const bf16* src = vv + bO + (size_t)tid * NSP + m0 + kc2 * 32;
        bf16* dst = sV + tid * LDA;
        *(v8bf*)(dst + 0)  = *(const v8bf*)(src + 0);
        *(v8bf*)(dst + 8)  = *(const v8bf*)(src + 8);
        *(v8bf*)(dst + 16) = *(const v8bf*)(src + 16);
        *(v8bf*)(dst + 24) = *(const v8bf*)(src + 24);
      }
#endif
      __syncthreads();
      v16bf aP = load_frag_a(sP + (wave & 3) * 16 * LDSS + kc2 * 32,
                             LDSS, lane);
      #pragma unroll
      for (int j = 0; j < 8; ++j) {
        v16bf bV = load_frag_b(sV + ((wave >> 2) * 128 + j * 16) * LDA,
                               LDA, lane);
        accO[j] = wmma_bf16(aP, bV, accO[j]);
      }
      __syncthreads();
    }
  }

  // ---- normalize and store o^T [B][N][C] bf16
  float linv[8];
  #pragma unroll
  for (int i = 0; i < 8; ++i)
    linv[i] = 1.f / sL[(wave & 3) * 16 + rhi * 8 + i];
  #pragma unroll
  for (int j = 0; j < 8; ++j) {
    int cc = (wave >> 2) * 128 + j * 16 + col;
    #pragma unroll
    for (int i = 0; i < 8; ++i) {
      int nn = n0 + (wave & 3) * 16 + rhi * 8 + i;
      oT[bO + (size_t)nn * CCH + cc] = (bf16)(accO[j][i] * linv[i]);
    }
  }
}

// ---------------------------------------------------------------------------
extern "C" void kernel_launch(void* const* d_in, const int* in_sizes, int n_in,
                              void* d_out, int out_size, void* d_ws,
                              size_t ws_size, hipStream_t stream) {
  const float* x  = (const float*)d_in[0];
  const float* gs = (const float*)d_in[1];
  const float* gb = (const float*)d_in[2];
  const float* wq = (const float*)d_in[3];
  const float* bq = (const float*)d_in[4];
  const float* wk = (const float*)d_in[5];
  const float* bk = (const float*)d_in[6];
  const float* wv = (const float*)d_in[7];
  const float* bv = (const float*)d_in[8];
  const float* wo = (const float*)d_in[9];
  const float* bo = (const float*)d_in[10];
  float* out = (float*)d_out;

  char* ws = (char*)d_ws;
  const size_t T = (size_t)BATCH * NC * sizeof(bf16);   // 16 MB per tensor
  bf16* hT  = (bf16*)(ws + 0 * T);
  bf16* qT  = (bf16*)(ws + 1 * T);
  bf16* kT  = (bf16*)(ws + 2 * T);
  bf16* vv  = (bf16*)(ws + 3 * T);
  bf16* oT  = (bf16*)(ws + 4 * T);
  bf16* wbf = (bf16*)(ws + 5 * T);                      // 4 * 65536 bf16

  cvt_w_kernel<<<1024, 256, 0, stream>>>(wq, wk, wv, wo, wbf);
  gn_kernel<<<BATCH * 32, 256, 0, stream>>>(x, gs, gb, hT);
  proj_gemm_kernel<<<dim3(16, 64, 3), 256, 0, stream>>>(wbf, bq, bk, bv, hT,
                                                        qT, kT, vv);
  attn_kernel<<<dim3(16, BATCH), 256, 0, stream>>>(qT, kT, vv, oT);
  outproj_gemm_kernel<<<dim3(16, 64), 256, 0, stream>>>(wbf + 3 * 65536, bo,
                                                        oT, x, out);
}
